// Multihead_Att_532575945174
// MI455X (gfx1250) — matrix operand order repaired and checked by reading, not compiled, not run on hardware
//
#include <hip/hip_runtime.h>

typedef unsigned short u16;
typedef unsigned int   u32;
typedef __attribute__((ext_vector_type(16))) __bf16 v16bf;
typedef __attribute__((ext_vector_type(8)))  float  v8f;

#define B_   2
#define QS_  2048
#define MS_  2048
#define KS_  4096
#define H_   1024
#define NH_  16
#define HD_  64
#define SCALE_ 0.125f

// ---------------------------------------------------------------- helpers ---
__device__ __forceinline__ u16 f2bf(float f) {
  union { float f; u32 u; } c; c.f = f;
  u32 r = c.u + 0x7FFFu + ((c.u >> 16) & 1u);   // round-to-nearest-even
  return (u16)(r >> 16);
}

struct alignas(16) U128 { u32 a, b, c, d; };    // trivially copyable 16B chunk

union Frag { U128 q[2]; v16bf v; };             // 32B = one WMMA A/B operand

__device__ __forceinline__ v8f wmma_bf16(const Frag& a, const Frag& b, v8f c) {
  return __builtin_amdgcn_wmma_f32_16x16x32_bf16(
      false, a.v, false, b.v, (short)0, c, false, false);
}

__device__ __forceinline__ v8f zero8() {
  v8f z;
#pragma unroll
  for (int i = 0; i < 8; ++i) z[i] = 0.f;
  return z;
}

// CDNA5 async DMA: global -> LDS, 16B per lane, tracked by ASYNCcnt.
// Low 32 bits of a flat LDS address are the LDS byte offset (aperture map).
__device__ __forceinline__ void async_load_b128(const void* gaddr, void* lds) {
  u32 l = (u32)(size_t)lds;
  asm volatile("global_load_async_to_lds_b128 %0, %1, off"
               :: "v"(l), "v"(gaddr) : "memory");
}
template <int N>
__device__ __forceinline__ void wait_async() {
  asm volatile("s_wait_asynccnt %0" :: "i"(N) : "memory");
}

// -------------------------------------------------------------- converters --
__global__ void build_c_kernel(const float* __restrict__ x,
                               const float* __restrict__ mem,
                               u16* __restrict__ cbf) {
  size_t idx = ((size_t)blockIdx.x * 256 + threadIdx.x) * 4;
  int row  = (int)(idx >> 10);
  int col  = (int)(idx & (H_ - 1));
  int bsel = row >> 12;              // / KS_
  int r    = row & (KS_ - 1);
  const float* src = (r < MS_)
      ? &mem[((size_t)bsel * MS_ + r) * H_ + col]
      : &x  [((size_t)bsel * QS_ + (r - MS_)) * H_ + col];
#pragma unroll
  for (int j = 0; j < 4; ++j) cbf[idx + j] = f2bf(src[j]);
}

__global__ void f32_to_bf16_kernel(const float* __restrict__ in,
                                   u16* __restrict__ out) {
  size_t idx = ((size_t)blockIdx.x * 256 + threadIdx.x) * 4;
#pragma unroll
  for (int j = 0; j < 4; ++j) out[idx + j] = f2bf(in[idx + j]);
}

// W: [K][N] fp32  ->  Wt: [N][K] bf16 (B-fragment friendly layout)
__global__ void transpose_w_kernel(const float* __restrict__ W,
                                   u16* __restrict__ Wt, int K, int N) {
  size_t idx = (size_t)blockIdx.x * 256 + threadIdx.x;
  int n = (int)(idx / K);
  int k = (int)(idx - (size_t)n * K);
  Wt[idx] = f2bf(W[(size_t)k * N + n]);
}

// ------------------------------------------------------------------- GEMM ---
// C[M,N] = A[M,K](bf16,row-major) * Bt[N,K](bf16, pre-transposed)
// Block tile 128x128, K-step 32, 8 waves each computing a 32x64 sub-tile.
// Double-buffered LDS fed by global_load_async_to_lds_b128.
#define BM  128
#define BN  128
#define BK  32
#define LDT 40      // LDS row pitch (u16) : 32 + 8 pad, keeps 16B alignment

__global__ __launch_bounds__(256) void gemm_bf16_kernel(
    const u16* __restrict__ A, const u16* __restrict__ Bt,
    int M, int N, int K, int mode,
    u16* __restrict__ obf, float* __restrict__ of,
    u16* __restrict__ kout, u16* __restrict__ vtout) {
  __shared__ __attribute__((aligned(16))) u16 sA[2][BM * LDT];
  __shared__ __attribute__((aligned(16))) u16 sB[2][BN * LDT];

  const int tid  = threadIdx.x;
  const int wave = tid >> 5, lane = tid & 31;
  const int half = lane >> 4, lm = lane & 15;
  const int wm = wave >> 1;                 // 0..3 : 32-row slab
  const int wn = wave & 1;                  // 0..1 : 64-col slab
  const int m0 = blockIdx.y * BM;
  const int n0 = blockIdx.x * BN;

  v8f acc[2][4];
#pragma unroll
  for (int mt = 0; mt < 2; ++mt)
#pragma unroll
    for (int nt = 0; nt < 4; ++nt) acc[mt][nt] = zero8();

  const int rA = tid >> 2;                  // 0..63
  const int cg = (tid & 3) * 8;             // 0,8,16,24 (u16 units)

  auto issue = [&](int kk, int buf) {       // 4 async 16B copies per thread
#pragma unroll
    for (int it = 0; it < 2; ++it) {
      int r = it * 64 + rA;
      async_load_b128(&A [(size_t)(m0 + r) * K + kk + cg], &sA[buf][r * LDT + cg]);
      async_load_b128(&Bt[(size_t)(n0 + r) * K + kk + cg], &sB[buf][r * LDT + cg]);
    }
  };

  issue(0, 0);
  const int nk = K / BK;
  for (int ik = 0; ik < nk; ++ik) {
    const int p = ik & 1;
    if (ik + 1 < nk) {                      // overlap next-tile DMA with WMMA
      issue((ik + 1) * BK, p ^ 1);
      wait_async<4>();                      // only cur-buffer copies must land
    } else {
      wait_async<0>();
    }
    __syncthreads();

    Frag a[2], b[4];
#pragma unroll
    for (int mt = 0; mt < 2; ++mt) {        // A-frag: M=lane%16, K split by half
      int row = wm * 32 + mt * 16 + lm;
      a[mt].q[0] = *(const U128*)&sA[p][row * LDT + half * 8];
      a[mt].q[1] = *(const U128*)&sA[p][row * LDT + 16 + half * 8];
    }
#pragma unroll
    for (int nt = 0; nt < 4; ++nt) {        // B-frag: N=lane%16, K split by half
      int col = wn * 64 + nt * 16 + lm;
      b[nt].q[0] = *(const U128*)&sB[p][col * LDT + half * 8];
      b[nt].q[1] = *(const U128*)&sB[p][col * LDT + 16 + half * 8];
    }
#pragma unroll
    for (int mt = 0; mt < 2; ++mt)
#pragma unroll
      for (int nt = 0; nt < 4; ++nt)
        acc[mt][nt] = wmma_bf16(a[mt], b[nt], acc[mt][nt]);
    __syncthreads();
  }

  // epilogue: C layout row = i + 8*(lane/16), col = lane%16
#pragma unroll
  for (int mt = 0; mt < 2; ++mt)
#pragma unroll
    for (int nt = 0; nt < 4; ++nt)
#pragma unroll
      for (int i = 0; i < 8; ++i) {
        int gm = m0 + wm * 32 + mt * 16 + i + half * 8;
        int gn = n0 + wn * 64 + nt * 16 + lm;
        float v = acc[mt][nt][i];
        if (mode == 0) {
          obf[(size_t)gm * N + gn] = f2bf(v);
        } else if (mode == 2) {
          of[(size_t)gm * N + gn] = v;
        } else {                            // KV mode: split K / transposed V
          if (gn < H_) {
            kout[(size_t)gm * H_ + gn] = f2bf(v);
          } else {
            int hh = (gn - H_) >> 6, dd = (gn - H_) & 63;
            int bb = gm >> 12, ks = gm & (KS_ - 1);
            vtout[(((size_t)(bb * NH_ + hh) * HD_) + dd) * KS_ + ks] = f2bf(v);
          }
        }
      }
}

// -------------------------------------------------------- flash attention ---
#define LKT 72      // LDS pitch for 64-wide tiles: 64 + 8 pad, 16B aligned

__global__ __launch_bounds__(256) void attn_kernel(
    const u16* __restrict__ qbf,   // [B*QS][H] bf16
    const u16* __restrict__ kbf,   // [B*KS][H] bf16
    const u16* __restrict__ vt,    // [B][NH][HD][KS] bf16
    u16* __restrict__ att) {       // [B*QS][H] bf16
  __shared__ __attribute__((aligned(16))) u16 sK[2][64 * LKT];    // [key][d]
  __shared__ __attribute__((aligned(16))) u16 sV[2][64 * LKT];    // [d][key]
  __shared__ __attribute__((aligned(16))) u16 sP[8][16 * LKT];    // per-wave P

  const int tid  = threadIdx.x;
  const int wave = tid >> 5, lane = tid & 31;
  const int half = lane >> 4, lm = lane & 15;

  const int qb = blockIdx.x & 15;                 // QS/128 = 16 q-blocks
  const int h  = (blockIdx.x >> 4) & (NH_ - 1);
  const int b  = blockIdx.x >> 8;
  const int q0 = qb * 128;
  const int qrow0 = q0 + wave * 16;               // this wave's q-tile

  // Q A-fragments (d = 0..63 -> two K-steps), resident for the whole sweep
  Frag aq[2];
  {
    const u16* qrow = qbf + ((size_t)(b * QS_ + qrow0 + lm)) * H_ + h * HD_;
#pragma unroll
    for (int ks = 0; ks < 2; ++ks) {
      aq[ks].q[0] = *(const U128*)&qrow[ks * 32 + half * 8];
      aq[ks].q[1] = *(const U128*)&qrow[ks * 32 + 16 + half * 8];
    }
  }

  v8f o[4];
#pragma unroll
  for (int dt = 0; dt < 4; ++dt) o[dt] = zero8();
  float mrow[8], lrow[8];
#pragma unroll
  for (int i = 0; i < 8; ++i) { mrow[i] = -__builtin_inff(); lrow[i] = 0.f; }

  auto issueKV = [&](int k0, int buf) {           // 4 async 16B copies / thread
#pragma unroll
    for (int it = 0; it < 2; ++it) {
      int chnk = it * 256 + tid;
      int r    = chnk >> 3;                       // 0..63
      int cgk  = (chnk & 7) * 8;                  // 0..56
      async_load_b128(&kbf[((size_t)(b * KS_ + k0 + r)) * H_ + h * HD_ + cgk],
                      &sK[buf][r * LKT + cgk]);
      async_load_b128(&vt[(((size_t)(b * NH_ + h) * HD_) + r) * KS_ + k0 + cgk],
                      &sV[buf][r * LKT + cgk]);
    }
  };

  const int kmax = q0 + MS_ + 128;                // causal-with-memory bound
  const int nchunks = kmax >> 6;
  issueKV(0, 0);

  for (int ic = 0; ic < nchunks; ++ic) {
    const int p  = ic & 1;
    const int k0 = ic * 64;
    if (ic + 1 < nchunks) {                       // prefetch next 64-key chunk
      issueKV(k0 + 64, p ^ 1);
      wait_async<4>();
    } else {
      wait_async<0>();
    }
    __syncthreads();

    if (k0 <= qrow0 + 15 + MS_) {                 // chunk not fully masked
      // S = Q * K^T  (4 N-tiles x 2 K-steps over d)
      v8f s[4];
#pragma unroll
      for (int nt = 0; nt < 4; ++nt) s[nt] = zero8();
#pragma unroll
      for (int nt = 0; nt < 4; ++nt)
#pragma unroll
        for (int ds = 0; ds < 2; ++ds) {
          Frag bk;
          const u16* kp = &sK[p][(nt * 16 + lm) * LKT + ds * 32 + half * 8];
          bk.q[0] = *(const U128*)kp;
          bk.q[1] = *(const U128*)(kp + 16);
          s[nt] = wmma_bf16(aq[ds], bk, s[nt]);
        }

      // online softmax per row (row = i + 8*half, col = lane%16)
#pragma unroll
      for (int i = 0; i < 8; ++i) {
        const int rg = qrow0 + i + half * 8;
        float pm = -__builtin_inff();
#pragma unroll
        for (int nt = 0; nt < 4; ++nt) {
          int cgl = k0 + nt * 16 + lm;
          float v = s[nt][i] * SCALE_;
          if (cgl > rg + MS_) v = -__builtin_inff();
          s[nt][i] = v;
          pm = fmaxf(pm, v);
        }
#pragma unroll
        for (int off = 1; off < 16; off <<= 1)
          pm = fmaxf(pm, __shfl_xor(pm, off, 32));
        float nm    = fmaxf(mrow[i], pm);
        float alpha = __expf(mrow[i] - nm);
        mrow[i] = nm;
        float ps = 0.f;
#pragma unroll
        for (int nt = 0; nt < 4; ++nt) {
          float pv = __expf(s[nt][i] - nm);
          ps += pv;
          sP[wave][(i + half * 8) * LKT + nt * 16 + lm] = f2bf(pv);
        }
#pragma unroll
        for (int off = 1; off < 16; off <<= 1)
          ps += __shfl_xor(ps, off, 32);
        lrow[i] = lrow[i] * alpha + ps;
#pragma unroll
        for (int dt = 0; dt < 4; ++dt) o[dt][i] *= alpha;
      }

      // O += P * V   (P A-frags re-read from this wave's LDS buffer)
      Frag ap[2];
#pragma unroll
      for (int ks = 0; ks < 2; ++ks) {
        const u16* pp = &sP[wave][lm * LKT + ks * 32 + half * 8];
        ap[ks].q[0] = *(const U128*)pp;
        ap[ks].q[1] = *(const U128*)(pp + 16);
      }
#pragma unroll
      for (int dt = 0; dt < 4; ++dt)
#pragma unroll
        for (int ks = 0; ks < 2; ++ks) {
          Frag bv;
          const u16* vp = &sV[p][(dt * 16 + lm) * LKT + ks * 32 + half * 8];
          bv.q[0] = *(const U128*)vp;
          bv.q[1] = *(const U128*)(vp + 16);
          o[dt] = wmma_bf16(ap[ks], bv, o[dt]);
        }
    }
    __syncthreads();                              // before buffer reuse
  }

  // normalize and store attended (bf16, natural [B*QS][H] layout)
#pragma unroll
  for (int i = 0; i < 8; ++i) {
    float inv = 1.f / lrow[i];
    size_t rowoff = ((size_t)(b * QS_ + qrow0 + i + half * 8)) * H_ + h * HD_;
#pragma unroll
    for (int dt = 0; dt < 4; ++dt)
      att[rowoff + dt * 16 + lm] = f2bf(o[dt][i] * inv);
  }
}

// ------------------------------------------------- residual + layer norm ----
__global__ __launch_bounds__(256) void ln_kernel(
    const float* __restrict__ x, const float* __restrict__ y,
    const float* __restrict__ g, const float* __restrict__ bb,
    float* __restrict__ out) {
  __shared__ float rs[8], rs2[8];
  const int row = blockIdx.x;
  const int tid = threadIdx.x;
  const int wave = tid >> 5, lane = tid & 31;
  const float* xr = x + (size_t)row * H_;
  const float* yr = y + (size_t)row * H_;
  float t[4], s = 0.f, s2 = 0.f;
#pragma unroll
  for (int j = 0; j < 4; ++j) {
    int idx = tid + j * 256;
    float v = xr[idx] + yr[idx];
    t[j] = v; s += v; s2 += v * v;
  }
#pragma unroll
  for (int off = 1; off < 32; off <<= 1) {
    s += __shfl_xor(s, off, 32);
    s2 += __shfl_xor(s2, off, 32);
  }
  if (lane == 0) { rs[wave] = s; rs2[wave] = s2; }
  __syncthreads();
  float ts = 0.f, ts2 = 0.f;
#pragma unroll
  for (int w = 0; w < 8; ++w) { ts += rs[w]; ts2 += rs2[w]; }
  const float mu  = ts * (1.f / H_);
  const float var = ts2 * (1.f / H_) - mu * mu;
  const float r   = rsqrtf(var + 1e-5f);
#pragma unroll
  for (int j = 0; j < 4; ++j) {
    int idx = tid + j * 256;
    out[(size_t)row * H_ + idx] = (t[j] - mu) * r * g[idx] + bb[idx];
  }
}

// ------------------------------------------------------------------ launch --
extern "C" void kernel_launch(void* const* d_in, const int* in_sizes, int n_in,
                              void* d_out, int out_size, void* d_ws, size_t ws_size,
                              hipStream_t stream) {
  (void)in_sizes; (void)n_in; (void)out_size; (void)ws_size;
  const float* x   = (const float*)d_in[0];
  const float* mem = (const float*)d_in[1];
  // d_in[2] = mask: recomputed analytically (k <= q + MS)
  const float* Wkv = (const float*)d_in[3];
  const float* Wq  = (const float*)d_in[4];
  const float* Wo  = (const float*)d_in[5];
  const float* lng = (const float*)d_in[6];
  const float* lnb = (const float*)d_in[7];
  float* out = (float*)d_out;

  char* ws = (char*)d_ws;
  size_t off = 0;
  auto alloc = [&](size_t bytes) -> char* {
    char* p = ws + off;
    off = (off + bytes + 255) & ~(size_t)255;
    return p;
  };
  u16*  cbf   = (u16*)alloc((size_t)B_ * KS_ * H_ * 2);          // concat bf16
  u16*  xbf   = (u16*)alloc((size_t)B_ * QS_ * H_ * 2);          // x bf16
  u16*  wkvT  = (u16*)alloc((size_t)2 * H_ * H_ * 2);            // Wkv^T bf16
  u16*  wqT   = (u16*)alloc((size_t)H_ * H_ * 2);                // Wq^T bf16
  u16*  woT   = (u16*)alloc((size_t)H_ * H_ * 2);                // Wo^T bf16
  u16*  kbf   = (u16*)alloc((size_t)B_ * KS_ * H_ * 2);          // K bf16
  u16*  vtt   = (u16*)alloc((size_t)B_ * NH_ * HD_ * KS_ * 2);   // V^T bf16
  u16*  qbf   = (u16*)alloc((size_t)B_ * QS_ * H_ * 2);          // Q bf16
  u16*  attbf = (u16*)alloc((size_t)B_ * QS_ * H_ * 2);          // attended bf16
  float* yf   = (float*)alloc((size_t)B_ * QS_ * H_ * 4);        // out-proj f32

  build_c_kernel<<<dim3((B_ * KS_ * H_) / 1024), 256, 0, stream>>>(x, mem, cbf);
  f32_to_bf16_kernel<<<dim3((B_ * QS_ * H_) / 1024), 256, 0, stream>>>(x, xbf);
  transpose_w_kernel<<<dim3((2 * H_ * H_) / 256), 256, 0, stream>>>(Wkv, wkvT, H_, 2 * H_);
  transpose_w_kernel<<<dim3((H_ * H_) / 256), 256, 0, stream>>>(Wq, wqT, H_, H_);
  transpose_w_kernel<<<dim3((H_ * H_) / 256), 256, 0, stream>>>(Wo, woT, H_, H_);

  // KV projection: [8192 x 1024] * [1024 x 2048] -> K + V^T
  gemm_bf16_kernel<<<dim3((2 * H_) / BN, (B_ * KS_) / BM), 256, 0, stream>>>(
      cbf, wkvT, B_ * KS_, 2 * H_, H_, 1, nullptr, nullptr, kbf, vtt);
  // Q projection: [4096 x 1024] * [1024 x 1024]
  gemm_bf16_kernel<<<dim3(H_ / BN, (B_ * QS_) / BM), 256, 0, stream>>>(
      xbf, wqT, B_ * QS_, H_, H_, 0, qbf, nullptr, nullptr, nullptr);

  // flash attention: (b, head, 128-q-row block)
  attn_kernel<<<dim3(B_ * NH_ * (QS_ / 128)), 256, 0, stream>>>(qbf, kbf, vtt, attbf);

  // output projection -> fp32
  gemm_bf16_kernel<<<dim3(H_ / BN, (B_ * QS_) / BM), 256, 0, stream>>>(
      attbf, woT, B_ * QS_, H_, H_, 2, nullptr, yf, nullptr, nullptr);

  // residual + layer norm
  ln_kernel<<<dim3(B_ * QS_), 256, 0, stream>>>(x, yf, lng, lnb, out);
}
